// RNN_58849641890526
// MI455X (gfx1250) — matrix operand order repaired and checked by reading, not compiled, not run on hardware
//
#include <hip/hip_runtime.h>

typedef unsigned short ushort_t;
typedef unsigned int   uint_t;
typedef __bf16 v16bf __attribute__((ext_vector_type(16)));
typedef float  v8f   __attribute__((ext_vector_type(8)));
typedef unsigned int u32x4 __attribute__((ext_vector_type(4)));
typedef int          i32x8 __attribute__((ext_vector_type(8)));
typedef int          i32x4 __attribute__((ext_vector_type(4)));

#define B_  64
#define S_  2048
#define I_  256
#define H_  512
#define NG  4        // batch groups (B/16)
#define NK  16       // column-slice WGs per group
#define LC  128      // local gate cols per WG (4 gates x 32)
#define TPB 256
#define AXP 272      // A_x row stride (ushorts): 256 + 16 pad  (136 dwords % 64 = 8)
#define AHP 520      // A_h row stride (ushorts): 512 + 8  pad  (260 dwords % 64 = 4), matches TDM pad

// ---- LDS layout for the recurrent kernel (byte offsets) ----
#define W_LDS_USH   (8 * 24 * 32 * 16)          // 98304 ushorts = 192KB
#define AX_OFF      (W_LDS_USH * 2)             // 196608
#define AH_OFF      (AX_OFF + 16 * AXP * 2)     // 205312
#define G_OFF       (AH_OFF + 16 * AHP * 2)     // 221952
#define C_OFF       (G_OFF + 16 * LC * 4)       // 230144
#define LDS_BYTES   (C_OFF + 16 * 32 * 4)       // 232192

union V16u { uint4 u[2]; v16bf v; };

__device__ __forceinline__ ushort_t f2bf(float f) {
  union { float f; uint_t u; } c; c.f = f;
  uint_t r = c.u + 0x7FFFu + ((c.u >> 16) & 1u);
  return (ushort_t)(r >> 16);
}
__device__ __forceinline__ float sigf(float x) { return 1.0f / (1.0f + __expf(-x)); }
__device__ __forceinline__ float tanh_fast(float x) {
  float e = __expf(-2.0f * fabsf(x));
  float t = (1.0f - e) / (1.0f + e);
  return x < 0.0f ? -t : t;
}

// ---------------- init: zero hs[t=0] and barrier state ----------------
__global__ void init_zero_k(ushort_t* hs, uint_t* bars) {
  int idx = blockIdx.x * TPB + threadIdx.x;     // 32768 threads
  ((uint_t*)hs)[idx] = 0;                       // 65536 ushorts = 2*B*H bf16 zeros
  if (idx < 64) bars[idx] = 0;
}

// ---------------- pack [Wx;Uh] slices into WMMA B-layout ----------------
__global__ void pack_w_k(const float* Wf, const float* Wi, const float* Wc, const float* Wo,
                         const float* Uf, const float* Ui, const float* Uc, const float* Uo,
                         ushort_t* packW) {
  int idx = blockIdx.x * TPB + threadIdx.x;     // 16*8*24*512 = 1,572,864
  int e    = idx & 15;
  int lane = (idx >> 4) & 31;
  int kk   = (idx >> 9) % 24;
  int rest = (idx >> 9) / 24;
  int nt   = rest & 7;
  int kwg  = rest >> 3;
  int n_local = nt * 16 + (lane & 15);          // 0..127
  int gate = n_local >> 5;
  int j    = (kwg << 5) + (n_local & 31);       // global h column
  int krow = kk * 32 + ((lane >= 16) ? 8 : 0) + (e & 7) + ((e >> 3) << 4);
  const float* Wg[4] = { Wf, Wi, Wc, Wo };
  const float* Ug[4] = { Uf, Ui, Uc, Uo };
  float v = (krow < I_) ? Wg[gate][(size_t)krow * H_ + j]
                        : Ug[gate][(size_t)(krow - I_) * H_ + j];
  packW[idx] = f2bf(v);
}

// ---------------- pack W_out [512,256] into WMMA B-layout ----------------
__global__ void pack_wout_k(const float* Wout, ushort_t* packWout) {
  int idx = blockIdx.x * TPB + threadIdx.x;     // 131072
  int e    = idx & 15;
  int lane = (idx >> 4) & 31;
  int kk   = (idx >> 9) & 15;
  int nt   = idx >> 13;
  int n    = nt * 16 + (lane & 15);
  int krow = kk * 32 + ((lane >= 16) ? 8 : 0) + (e & 7) + ((e >> 3) << 4);
  packWout[idx] = f2bf(Wout[(size_t)krow * I_ + n]);
}

// ---------------- persistent recurrent kernel ----------------
__global__ void lstm_persistent_k(const float* __restrict__ X,
                                  const ushort_t* __restrict__ packW,
                                  const float* __restrict__ bf_, const float* __restrict__ bi_,
                                  const float* __restrict__ bc_, const float* __restrict__ bo_,
                                  ushort_t* __restrict__ hs,
                                  uint_t* __restrict__ bars) {
  extern __shared__ char smem[];
  ushort_t* Wls = (ushort_t*)smem;
  ushort_t* Ax  = (ushort_t*)(smem + AX_OFF);
  ushort_t* Ah  = (ushort_t*)(smem + AH_OFF);
  float*    Gls = (float*)(smem + G_OFF);
  float*    Cls = (float*)(smem + C_OFF);

  const int tid  = threadIdx.x;
  const int lane = tid & 31;
  const int w    = tid >> 5;
  const int g    = blockIdx.x >> 4;
  const int kwg  = blockIdx.x & 15;
  const int b0   = g << 4;
  uint_t* bar = bars + g * 16;                  // 64B apart per group

  // load this WG's weight slice into LDS (once)
  {
    const uint4* src = (const uint4*)(packW + (size_t)kwg * W_LDS_USH);
    uint4* dst = (uint4*)Wls;
    for (int i = tid; i < W_LDS_USH / 8; i += TPB) dst[i] = src[i];
  }
  for (int i = tid; i < 512; i += TPB) Cls[i] = 0.0f;

  // hoist gate biases for this thread's two (m,jj) elements out of the t-loop
  const int e0 = tid, e1 = tid + TPB;
  const int j0 = (kwg << 5) + (e0 & 31), j1 = (kwg << 5) + (e1 & 31);
  const float bf0 = bf_[j0], bi0 = bi_[j0], bc0 = bc_[j0], bo0 = bo_[j0];
  const float bf1 = bf_[j1], bi1 = bi_[j1], bc1 = bc_[j1], bo1 = bo_[j1];

  // stage X(t=0) into A_x
  for (int it = 0; it < 4; ++it) {
    int idx = tid + it * TPB;
    int m  = idx >> 6;
    int c4 = (idx & 63) << 2;
    float4 xv = *(const float4*)(X + ((size_t)(b0 + m) * S_ + 0) * I_ + c4);
    uint2 p;
    p.x = (uint_t)f2bf(xv.x) | ((uint_t)f2bf(xv.y) << 16);
    p.y = (uint_t)f2bf(xv.z) | ((uint_t)f2bf(xv.w) << 16);
    *(uint2*)(Ax + m * AXP + c4) = p;
  }

  const int abase  = (lane >> 4) << 3;          // A-layout K base: 0 or 8
  const int ax_row = (lane & 15) * AXP + abase;
  const int ah_row = (lane & 15) * AHP + abase;
  const unsigned long long hs_base = (unsigned long long)(uintptr_t)hs
                                   + (unsigned long long)b0 * (H_ * 2);

  for (int t = 0; t < S_; ++t) {
    // ---- TDM: DMA h_{t-1} tile [16 x 512] bf16 global -> LDS with row padding ----
    if (w == 0) {
      unsigned long long ga = hs_base + (unsigned long long)t * (B_ * H_ * 2);
      u32x4 g0 = { 1u,                           // count = 1
                   (uint_t)AH_OFF,               // lds_addr (byte offset in our allocation)
                   (uint_t)ga,                   // global_addr[31:0]
                   (uint_t)(ga >> 32) | (2u << 30) };  // global_addr[56:32] | type=2
      i32x8 g1 = { (1 << 16) | (1 << 20) | (7 << 22) | (3 << 25), // 2B, pad: every 256 dw add 4 dw
                   (int)(512u << 16),            // tensor_dim0 = 512 (low 16)
                   (int)(16u << 16),             // tensor_dim0 hi=0 | tensor_dim1 = 16 (low 16)
                   (int)(512u << 16),            // tensor_dim1 hi=0 | tile_dim0 = 512
                   16,                           // tile_dim1 = 16, tile_dim2 = 0
                   512,                          // tensor_dim0_stride = 512 (low 32)
                   0, 0 };
      i32x4 gz4 = { 0, 0, 0, 0 };
      i32x8 gz8 = { 0, 0, 0, 0, 0, 0, 0, 0 };
      __builtin_amdgcn_tensor_load_to_lds(g0, g1, gz4, gz4, gz8, 0);
      __builtin_amdgcn_s_wait_tensorcnt(0);
    }
    __syncthreads();

    // ---- pipelined GEMM: [16 x 768] x [768 x 128]; wave w owns n-tile w ----
    v8f acc = {0.f, 0.f, 0.f, 0.f, 0.f, 0.f, 0.f, 0.f};
    {
#define LOAD_A(KK, DST) do { \
      const ushort_t* _ap = ((KK) < 8) ? (Ax + ax_row + (KK) * 32) \
                                       : (Ah + ah_row + ((KK) - 8) * 32); \
      (DST).u[0] = *(const uint4*)_ap; (DST).u[1] = *(const uint4*)(_ap + 16); } while (0)
#define LOAD_B(KK, DST) do { \
      const ushort_t* _bp = Wls + (((w * 24 + (KK)) << 5) + lane) * 16; \
      (DST).u[0] = *(const uint4*)_bp; (DST).u[1] = *(const uint4*)(_bp + 8); } while (0)
      V16u a_c, b_c, a_n, b_n;
      LOAD_A(0, a_c); LOAD_B(0, b_c);
      #pragma unroll
      for (int kk = 0; kk < 24; ++kk) {
        if (kk < 23) { LOAD_A(kk + 1, a_n); LOAD_B(kk + 1, b_n); }
        acc = __builtin_amdgcn_wmma_f32_16x16x32_bf16(
            false, a_c.v, false, b_c.v, (short)0, acc, false, false);
        a_c = a_n; b_c = b_n;
      }
#undef LOAD_A
#undef LOAD_B
    }
    // spill gate tile to LDS (D-layout: n = lane&15, m = r + 8*(lane>=16))
    {
      int n  = (w << 4) + (lane & 15);
      int mb = (lane >> 4) << 3;
      #pragma unroll
      for (int r = 0; r < 8; ++r) Gls[(mb + r) * LC + n] = acc[r];
    }
    __syncthreads();

    // ---- elementwise gates (2 elems/thread), biases pre-hoisted ----
    {
      int m  = e0 >> 5, jj = e0 & 31;
      float gf = Gls[m * LC + jj]      + bf0;
      float gi = Gls[m * LC + 32 + jj] + bi0;
      float gc = Gls[m * LC + 64 + jj] + bc0;
      float go = Gls[m * LC + 96 + jj] + bo0;
      float cn = sigf(gf) * Cls[e0] + sigf(gi) * tanh_fast(gc);
      Cls[e0] = cn;
      float hn = sigf(go) * tanh_fast(cn);
      hs[(size_t)(t + 1) * (B_ * H_) + (size_t)(b0 + m) * H_ + j0] = f2bf(hn);
    }
    {
      int m  = e1 >> 5, jj = e1 & 31;
      float gf = Gls[m * LC + jj]      + bf1;
      float gi = Gls[m * LC + 32 + jj] + bi1;
      float gc = Gls[m * LC + 64 + jj] + bc1;
      float go = Gls[m * LC + 96 + jj] + bo1;
      float cn = sigf(gf) * Cls[e1] + sigf(gi) * tanh_fast(gc);
      Cls[e1] = cn;
      float hn = sigf(go) * tanh_fast(cn);
      hs[(size_t)(t + 1) * (B_ * H_) + (size_t)(b0 + m) * H_ + j1] = f2bf(hn);
    }

    // ---- overlap: stage X(t+1) while h writes drain / barrier spins ----
    if (t + 1 < S_) {
      for (int it = 0; it < 4; ++it) {
        int idx = tid + it * TPB;
        int m  = idx >> 6;
        int c4 = (idx & 63) << 2;
        float4 xv = *(const float4*)(X + ((size_t)(b0 + m) * S_ + (t + 1)) * I_ + c4);
        uint2 p;
        p.x = (uint_t)f2bf(xv.x) | ((uint_t)f2bf(xv.y) << 16);
        p.y = (uint_t)f2bf(xv.z) | ((uint_t)f2bf(xv.w) << 16);
        *(uint2*)(Ax + m * AXP + c4) = p;
      }
    }

    // ---- per-group sense-reversing barrier (16 WGs exchange h slices) ----
    __threadfence();
    __syncthreads();
    if (tid == 0) {
      uint_t gen = __hip_atomic_load(&bar[1], __ATOMIC_ACQUIRE, __HIP_MEMORY_SCOPE_AGENT);
      uint_t v = __hip_atomic_fetch_add(&bar[0], 1u, __ATOMIC_ACQ_REL, __HIP_MEMORY_SCOPE_AGENT);
      if (v == NK - 1) {
        __hip_atomic_store(&bar[0], 0u, __ATOMIC_RELAXED, __HIP_MEMORY_SCOPE_AGENT);
        __hip_atomic_fetch_add(&bar[1], 1u, __ATOMIC_RELEASE, __HIP_MEMORY_SCOPE_AGENT);
      } else {
        while (__hip_atomic_load(&bar[1], __ATOMIC_ACQUIRE, __HIP_MEMORY_SCOPE_AGENT) == gen)
          __builtin_amdgcn_s_sleep(1);
      }
    }
    __syncthreads();
  }
}

// ---------------- output GEMM: out[b,s,:] = hs[s+1,b,:] @ W_out + b_out ----------------
__global__ void out_gemm_k(const ushort_t* __restrict__ hs,
                           const ushort_t* __restrict__ packWout,
                           const float* __restrict__ bout,
                           float* __restrict__ out) {
  const int s    = blockIdx.x;
  const int tid  = threadIdx.x;
  const int lane = tid & 31;
  const int w    = tid >> 5;
  const int mt   = w & 3;
  const int nh   = w >> 2;
  const int abase = (lane >> 4) << 3;

  v8f acc[8];
  #pragma unroll
  for (int nt = 0; nt < 8; ++nt) acc[nt] = (v8f){0.f,0.f,0.f,0.f,0.f,0.f,0.f,0.f};

  const ushort_t* Arow = hs + ((size_t)(s + 1) * B_ + mt * 16 + (lane & 15)) * H_ + abase;
  V16u a_c, a_n;
  a_c.u[0] = *(const uint4*)Arow;
  a_c.u[1] = *(const uint4*)(Arow + 16);
  for (int kk = 0; kk < 16; ++kk) {
    // batch all 8 B-tile loads (independent, stay in flight together)
    V16u bms[8];
    #pragma unroll
    for (int nt = 0; nt < 8; ++nt) {
      const ushort_t* bp = packWout + ((((nh * 8 + nt) * 16 + kk) << 5) + lane) * 16;
      bms[nt].u[0] = *(const uint4*)bp;
      bms[nt].u[1] = *(const uint4*)(bp + 8);
    }
    if (kk < 15) {
      const ushort_t* ap = Arow + (kk + 1) * 32;
      a_n.u[0] = *(const uint4*)ap;
      a_n.u[1] = *(const uint4*)(ap + 16);
    }
    #pragma unroll
    for (int nt = 0; nt < 8; ++nt)
      acc[nt] = __builtin_amdgcn_wmma_f32_16x16x32_bf16(
          false, a_c.v, false, bms[nt].v, (short)0, acc[nt], false, false);
    a_c = a_n;
  }
  const int mb = (lane >> 4) << 3;
  #pragma unroll
  for (int nt = 0; nt < 8; ++nt) {
    int i = nh * 128 + nt * 16 + (lane & 15);
    float bias = bout[i];
    #pragma unroll
    for (int r = 0; r < 8; ++r) {
      int b = mt * 16 + mb + r;
      out[((size_t)b * S_ + s) * I_ + i] = acc[nt][r] + bias;
    }
  }
}

// ---------------- host ----------------
extern "C" void kernel_launch(void* const* d_in, const int* in_sizes, int n_in,
                              void* d_out, int out_size, void* d_ws, size_t ws_size,
                              hipStream_t stream) {
  const float* X    = (const float*)d_in[0];
  const float* Wf   = (const float*)d_in[1];
  const float* Wi   = (const float*)d_in[2];
  const float* Wc   = (const float*)d_in[3];
  const float* Wo   = (const float*)d_in[4];
  const float* Uf   = (const float*)d_in[5];
  const float* Ui   = (const float*)d_in[6];
  const float* Uc   = (const float*)d_in[7];
  const float* Uo   = (const float*)d_in[8];
  const float* bf   = (const float*)d_in[9];
  const float* bi   = (const float*)d_in[10];
  const float* bc   = (const float*)d_in[11];
  const float* bo   = (const float*)d_in[12];
  const float* Wout = (const float*)d_in[13];
  const float* bout = (const float*)d_in[14];
  float* out = (float*)d_out;

  char* ws = (char*)d_ws;
  ushort_t* hs       = (ushort_t*)ws;                               // (S+1)*B*H bf16
  size_t hs_bytes    = (size_t)(S_ + 1) * B_ * H_ * 2;              // 134,283,264
  ushort_t* packW    = (ushort_t*)(ws + hs_bytes);                  // 3,145,728 B
  size_t pw_bytes    = (size_t)NK * W_LDS_USH * 2;
  ushort_t* packWout = (ushort_t*)(ws + hs_bytes + pw_bytes);       // 262,144 B
  size_t pwo_bytes   = (size_t)16 * 16 * 32 * 16 * 2;
  uint_t* bars       = (uint_t*)(ws + hs_bytes + pw_bytes + pwo_bytes);

  init_zero_k<<<128, TPB, 0, stream>>>(hs, bars);
  pack_w_k<<<6144, TPB, 0, stream>>>(Wf, Wi, Wc, Wo, Uf, Ui, Uc, Uo, packW);
  pack_wout_k<<<512, TPB, 0, stream>>>(Wout, packWout);
  lstm_persistent_k<<<NG * NK, TPB, LDS_BYTES, stream>>>(X, packW, bf, bi, bc, bo, hs, bars);
  out_gemm_k<<<S_, TPB, 0, stream>>>(hs, packWout, bout, out);
}
